// DGCNN_53102975648388
// MI455X (gfx1250) — matrix operand order repaired and checked by reading, not compiled, not run on hardware
//
#include <hip/hip_runtime.h>
#include <hip/hip_bf16.h>
#include <stdint.h>

typedef _Float16 f16;
typedef _Float16 v8h  __attribute__((ext_vector_type(8)));
typedef _Float16 v16h __attribute__((ext_vector_type(16)));
typedef float    v8f  __attribute__((ext_vector_type(8)));

#define NB      4
#define NPTS    2048
#define KNN     20
#define NKCOLS  (NPTS*KNN)      // 40960 edge-feature columns per batch
#define NEG_INF (-3.0e38f)
#define W5K     512

#if defined(__has_builtin)
#if __has_builtin(__builtin_amdgcn_tensor_load_to_lds) && \
    __has_builtin(__builtin_amdgcn_s_wait_tensorcnt)
#define HAVE_TDM 1
#endif
#endif

#ifdef HAVE_TDM
typedef unsigned int u32x4 __attribute__((ext_vector_type(4)));
typedef int          i32x8 __attribute__((ext_vector_type(8)));
typedef int          i32x4 __attribute__((ext_vector_type(4)));
#endif

typedef __attribute__((address_space(3))) f16  lf16;
typedef __attribute__((address_space(3))) v8h  lv8h;

static inline int cdiv(int a, int b) { return (a + b - 1) / b; }

// ---------------------------------------------------------------------------
// WMMA wrapper: D = A(16x32 f16) * B(32x16 f16) + C(16x16 f32)
// ---------------------------------------------------------------------------
__device__ __forceinline__ v8f wmma_f32_f16(v16h a, v16h b, v8f c) {
  return __builtin_amdgcn_wmma_f32_16x16x32_f16(
      /*neg_a=*/false, a, /*neg_b=*/false, b,
      /*c_mod=*/(short)0, c, /*reuse_a=*/false, /*reuse_b=*/false);
}

// A fragment (16x32 f16): lane owns row M=lane&15, K = h*16 + 0..15 contiguous.
__device__ __forceinline__ v16h ld_afrag(const f16* __restrict__ rowp,
                                         int kb, int h) {
  v8h lo = *(const v8h*)(rowp + kb + h * 16);
  v8h hi = *(const v8h*)(rowp + kb + h * 16 + 8);
  return __builtin_shufflevector(lo, hi, 0, 1, 2, 3, 4, 5, 6, 7,
                                 8, 9, 10, 11, 12, 13, 14, 15);
}

// Same, from LDS (forces ds_load_b128)
__device__ __forceinline__ v16h ld_afrag_lds(const lf16* rowp, int kb, int h) {
  v8h lo = *(const lv8h*)(rowp + kb + h * 16);
  v8h hi = *(const lv8h*)(rowp + kb + h * 16 + 8);
  return __builtin_shufflevector(lo, hi, 0, 1, 2, 3, 4, 5, 6, 7,
                                 8, 9, 10, 11, 12, 13, 14, 15);
}

// B fragment (32x16 f16): lane owns col N=lane&15, VGPR j holds K=h*16+2j,+1.
__device__ __forceinline__ v16h ld_bfrag(const f16* __restrict__ rowp,
                                         int kb, int h) {
  return *(const v16h*)(rowp + kb + h * 16);
}

// ---------------------------------------------------------------------------
// BN folding: scale = g*rsqrt(v+eps), shift = b - m*scale
// ---------------------------------------------------------------------------
__global__ __launch_bounds__(256) void bn_prep_kernel(
    const float* __restrict__ bn, float* __restrict__ scale,
    float* __restrict__ shift, int c) {
  int i = blockIdx.x * blockDim.x + threadIdx.x;
  if (i >= c) return;
  float g  = bn[i];
  float be = bn[c + i];
  float m  = bn[2*c + i];
  float v  = bn[3*c + i];
  float sc = g * rsqrtf(v + 1e-5f);
  scale[i] = sc;
  shift[i] = be - m * sc;
}

// Weight (O x Cw fp32) -> (O x Kpad f16), zero padded K.
__global__ __launch_bounds__(256) void cvt_w_kernel(
    const float* __restrict__ W, f16* __restrict__ Wh, int O, int Cw, int Kpad) {
  int i = blockIdx.x * blockDim.x + threadIdx.x;
  if (i >= O * Kpad) return;
  int o = i / Kpad, c = i % Kpad;
  Wh[i] = (c < Cw) ? (f16)W[o * Cw + c] : (f16)0.f;
}

// Point-major f16 copy of features: xhT[b][n][Cpad], zero pad channels >= Cin.
__global__ __launch_bounds__(256) void cvt_xh_kernel(
    const float* __restrict__ xin0, long bstride,
    f16* __restrict__ xhT, int Cin, int Cpad) {
  int i = blockIdx.x * blockDim.x + threadIdx.x;
  int tot = NB * NPTS * Cpad;
  if (i >= tot) return;
  int b = i / (NPTS * Cpad);
  int rm = i % (NPTS * Cpad);
  int n = rm / Cpad, c = rm % Cpad;
  xhT[i] = (c < Cin) ? (f16)xin0[(size_t)b * bstride + (size_t)c * NPTS + n]
                     : (f16)0.f;
}

// Point-major f16 copy of xcat: xcatT[b][n][512]
__global__ __launch_bounds__(256) void cvt_xcatT_kernel(
    const float* __restrict__ xcat, f16* __restrict__ xT) {
  int i = blockIdx.x * blockDim.x + threadIdx.x;
  int tot = NB * NPTS * 512;
  if (i >= tot) return;
  int b = i / (NPTS * 512);
  int rm = i % (NPTS * 512);
  int n = rm / 512, c = rm % 512;
  xT[i] = (f16)xcat[((size_t)b * 512 + c) * NPTS + n];
}

// Squared norms xx[b][n] (fp32 exact)
__global__ __launch_bounds__(256) void sqnorm_kernel(
    const float* __restrict__ xin0, long bstride,
    float* __restrict__ xx, int Cin) {
  int i = blockIdx.x * blockDim.x + threadIdx.x;
  if (i >= NB * NPTS) return;
  int b = i / NPTS, n = i % NPTS;
  const float* p = xin0 + (size_t)b * bstride + n;
  float s = 0.f;
  for (int c = 0; c < Cin; ++c) { float v = p[(size_t)c * NPTS]; s += v * v; }
  xx[i] = s;
}

// ---------------------------------------------------------------------------
// pd[b][n][m] = 2*dot(x_n,x_m) - xx[n] - xx[m].  Gram via WMMA; each wave
// computes a 16x32 strip (A fragment reused across two column tiles).
// ---------------------------------------------------------------------------
__global__ __launch_bounds__(128) void gram_pd_kernel(
    const f16* __restrict__ xhT, const float* __restrict__ xx,
    float* __restrict__ pd, int Cpad) {
  int gtid = blockIdx.x * blockDim.x + threadIdx.x;
  int wave = gtid >> 5, lane = gtid & 31;
  const int T = NPTS / 16, MP = T / 2;   // 128 row tiles x 64 col-tile pairs
  int tpb = T * MP;
  int b = wave / tpb;
  if (b >= NB) return;
  int rem = wave % tpb;
  int nt = rem / MP, mp = rem % MP;
  int h = lane >> 4, r = lane & 15;
  int nrow = nt * 16 + r;
  int mc0 = mp * 32 + r, mc1 = mc0 + 16;
  const f16* xb = xhT + (size_t)b * NPTS * Cpad;
  const f16* arow  = xb + (size_t)nrow * Cpad;
  const f16* brow0 = xb + (size_t)mc0 * Cpad;
  const f16* brow1 = xb + (size_t)mc1 * Cpad;
  v8f acc0 = {}, acc1 = {};
  for (int kb = 0; kb < Cpad; kb += 32) {
    v16h a  = ld_afrag(arow, kb, h);
    v16h b0 = ld_bfrag(brow0, kb, h);
    v16h b1 = ld_bfrag(brow1, kb, h);
    acc0 = wmma_f32_f16(a, b0, acc0);
    acc1 = wmma_f32_f16(a, b1, acc1);
  }
  const float* xxb = xx + b * NPTS;
#pragma unroll
  for (int jj = 0; jj < 8; ++jj) {      // C/D: VGPR j -> M=j / j+8
    int nr = nt * 16 + jj + (h ? 8 : 0);
    float* prow = pd + ((size_t)(b * NPTS) + nr) * NPTS;
    prow[mc0] = 2.f * acc0[jj] - xxb[nr] - xxb[mc0];
    prow[mc1] = 2.f * acc1[jj] - xxb[nr] - xxb[mc1];
  }
}

// ---------------------------------------------------------------------------
// Top-20 per row: per-lane sorted candidate lists, wave32 shuffle-merge.
// ---------------------------------------------------------------------------
__global__ __launch_bounds__(128) void topk_kernel(
    const float* __restrict__ pd, int* __restrict__ idxout) {
  int gtid = blockIdx.x * blockDim.x + threadIdx.x;
  int wave = gtid >> 5, lane = gtid & 31;
  if (wave >= NB * NPTS) return;
  int b = wave / NPTS, n = wave % NPTS;
  const float* row = pd + ((size_t)b * NPTS + n) * NPTS;
  float d[KNN]; int id[KNN];
#pragma unroll
  for (int t = 0; t < KNN; ++t) { d[t] = NEG_INF; id[t] = 0x7fffffff; }
  for (int m = lane; m < NPTS; m += 32) {
    float v = row[m];
    if (v > d[KNN-1] || (v == d[KNN-1] && m < id[KNN-1])) {
      d[KNN-1] = v; id[KNN-1] = m;
#pragma unroll
      for (int t = KNN-1; t > 0; --t) {
        if (d[t] > d[t-1] || (d[t] == d[t-1] && id[t] < id[t-1])) {
          float tv = d[t]; d[t] = d[t-1]; d[t-1] = tv;
          int   ti = id[t]; id[t] = id[t-1]; id[t-1] = ti;
        }
      }
    }
  }
  int p = 0;
  float cv = d[0]; int ci = id[0];
  int* out = idxout + ((size_t)b * NPTS + n) * KNN;
  for (int s = 0; s < KNN; ++s) {
    float bv = cv; int bi = ci; int bl = lane;
#pragma unroll
    for (int off = 16; off > 0; off >>= 1) {
      float ov = __shfl_xor(bv, off, 32);
      int   oi = __shfl_xor(bi, off, 32);
      int   ol = __shfl_xor(bl, off, 32);
      if (ov > bv || (ov == bv && oi < bi)) { bv = ov; bi = oi; bl = ol; }
    }
    if (lane == 0) out[s] = bi;
    if (lane == bl) {                 // winner advances its list pointer
      ++p;
      cv = NEG_INF; ci = 0x7fffffff;
#pragma unroll
      for (int t = 1; t < KNN; ++t) if (p == t) { cv = d[t]; ci = id[t]; }
    }
  }
}

// ---------------------------------------------------------------------------
// Edge conv GEMM (gather-fused) + BN + leaky.  Each wave: one gathered
// B fragment reused across TWO output-row tiles (two wmma per k-step).
// Edge feature channels: [0,Cin)=nb-ctr, [Cin,2Cin)=ctr, rest 0.
// ---------------------------------------------------------------------------
__global__ __launch_bounds__(128) void edge_gemm_kernel(
    const f16* __restrict__ Wh, const f16* __restrict__ xhT, int Cstride,
    const int* __restrict__ idxb,
    const float* __restrict__ bnscale, const float* __restrict__ bnshift,
    float* __restrict__ fout, int O, int Cin, int Kpad) {
  int gtid = blockIdx.x * blockDim.x + threadIdx.x;
  int wave = gtid >> 5, lane = gtid & 31;
  const int colT = NKCOLS / 16;     // 2560
  int tpb = (O / 32) * colT;
  int b = wave / tpb;
  if (b >= NB) return;
  int rem = wave % tpb;
  int op = rem / colT, ct = rem % colT;
  int h = lane >> 4, r = lane & 15;
  int col = ct * 16 + r;
  int n = col / KNN, kk = col % KNN;
  int j = idxb[((size_t)b * NPTS + n) * KNN + kk];
  const f16* xb = xhT + (size_t)b * NPTS * Cstride;
  const f16* pj = xb + (size_t)j * Cstride;
  const f16* pn = xb + (size_t)n * Cstride;
  const f16* wr0 = Wh + (size_t)(op * 32 + r) * Kpad;
  const f16* wr1 = wr0 + (size_t)16 * Kpad;
  v8f acc0 = {}, acc1 = {};
  for (int kb = 0; kb < Kpad; kb += 32) {
    __builtin_prefetch(wr0 + kb + 32, 0, 1);   // global_prefetch_b8
    v16h a0 = ld_afrag(wr0, kb, h);
    v16h a1 = ld_afrag(wr1, kb, h);
    v16h bb;
    int c0 = kb + h * 16;
#pragma unroll
    for (int e = 0; e < 16; ++e) {
      int c = c0 + e;
      f16 v;
      if (c < Cin)          v = pj[c] - pn[c];
      else if (c < 2 * Cin) v = pn[c - Cin];
      else                  v = (f16)0.f;
      bb[e] = v;
    }
    acc0 = wmma_f32_f16(a0, bb, acc0);
    acc1 = wmma_f32_f16(a1, bb, acc1);
  }
#pragma unroll
  for (int jj = 0; jj < 8; ++jj) {
    int o0 = op * 32 + jj + (h ? 8 : 0);
    int o1 = o0 + 16;
    float v0 = acc0[jj] * bnscale[o0] + bnshift[o0];
    float v1 = acc1[jj] * bnscale[o1] + bnshift[o1];
    v0 = v0 > 0.f ? v0 : 0.2f * v0;            // leaky 0.2
    v1 = v1 > 0.f ? v1 : 0.2f * v1;
    fout[((size_t)(b * O + o0)) * NKCOLS + col] = v0;
    fout[((size_t)(b * O + o1)) * NKCOLS + col] = v1;
  }
}

// ---------------------------------------------------------------------------
// W5 GEMM (O=1024, K=512) with the 32-row weight tile staged into LDS by the
// Tensor Data Mover (TENSOR_LOAD_TO_LDS + s_wait_tensorcnt); all 4 waves of
// the workgroup then read A fragments from LDS (ds_load_b128) while streaming
// B fragments from global.  BN + leaky fused in the epilogue.
// ---------------------------------------------------------------------------
__global__ __launch_bounds__(128) void plain_gemm_tdm_kernel(
    const f16* __restrict__ Wh, const f16* __restrict__ xT,
    const float* __restrict__ bnscale, const float* __restrict__ bnshift,
    float* __restrict__ out) {
  const int O = 1024, K = W5K, Ncols = NPTS;
  __shared__ f16 wsm[32 * W5K];              // 32 KB: one 32-row W tile
  int tid = threadIdx.x;
  int waveId = tid >> 5, lane = tid & 31;
  const int ctgPerB = Ncols / 16 / 4;        // 32 groups of 4 col tiles
  const int blocksPerB = (O / 32) * ctgPerB; // 1024
  int b   = blockIdx.x / blocksPerB;
  int rm  = blockIdx.x % blocksPerB;
  int op  = rm / ctgPerB;
  int ctg = rm % ctgPerB;
  int ct  = ctg * 4 + waveId;

  const f16* wtile = Wh + (size_t)(op * 32) * K;
#ifdef HAVE_TDM
  if (waveId == 0) {                         // one TDM issue per workgroup
    unsigned lds = (unsigned)(uintptr_t)(lf16*)&wsm[0];
    unsigned long long ga = (unsigned long long)(uintptr_t)wtile;
    u32x4 g0; i32x8 g1; i32x4 g2, g3; i32x8 g4;
    g0[0] = 1u;                                            // count=1, user D#
    g0[1] = lds;                                           // lds_addr (bytes)
    g0[2] = (unsigned)(ga & 0xffffffffu);                  // global_addr lo
    g0[3] = (unsigned)((ga >> 32) & 0x01ffffffu) | (2u << 30); // hi | type=2
    g1[0] = (int)(1u << 16);            // data_size=1 (2 bytes), no multicast
    g1[1] = (int)((unsigned)(K & 0xffff) << 16);  // tensor_dim0[15:0]
    g1[2] = (int)((unsigned)(K >> 16) | (32u << 16)); // td0 hi | tensor_dim1 lo
    g1[3] = (int)((unsigned)K << 16);   // tensor_dim1 hi=0 | tile_dim0=K
    g1[4] = 32;                         // tile_dim1=32, tile_dim2=0
    g1[5] = K;                          // tensor_dim0_stride lo32
    g1[6] = 0;                          // stride0 hi | stride1 lo (unused, 2D)
    g1[7] = 0;
    g2[0] = g2[1] = g2[2] = g2[3] = 0;
    g3[0] = g3[1] = g3[2] = g3[3] = 0;
    g4[0] = g4[1] = g4[2] = g4[3] = 0;
    g4[4] = g4[5] = g4[6] = g4[7] = 0;
    __builtin_amdgcn_tensor_load_to_lds(g0, g1, g2, g3, g4, 0);
    __builtin_amdgcn_s_wait_tensorcnt(0);
  }
  __syncthreads();
#else
  for (int i = tid; i < 32 * K / 8; i += 128)
    ((v8h*)wsm)[i] = ((const v8h*)wtile)[i];
  __syncthreads();
#endif

  int h = lane >> 4, r = lane & 15;
  int col = ct * 16 + r;
  const f16* xrow = xT + ((size_t)b * Ncols + col) * K;
  const lf16* wl0 = (const lf16*)&wsm[(size_t)r * K];
  const lf16* wl1 = (const lf16*)&wsm[(size_t)(16 + r) * K];
  v8f acc0 = {}, acc1 = {};
  for (int kb = 0; kb < K; kb += 32) {
    __builtin_prefetch(xrow + kb + 32, 0, 1);
    v16h a0 = ld_afrag_lds(wl0, kb, h);
    v16h a1 = ld_afrag_lds(wl1, kb, h);
    v16h bb = ld_bfrag(xrow, kb, h);
    acc0 = wmma_f32_f16(a0, bb, acc0);
    acc1 = wmma_f32_f16(a1, bb, acc1);
  }
#pragma unroll
  for (int jj = 0; jj < 8; ++jj) {
    int o0 = op * 32 + jj + (h ? 8 : 0);
    int o1 = o0 + 16;
    float v0 = acc0[jj] * bnscale[o0] + bnshift[o0];
    float v1 = acc1[jj] * bnscale[o1] + bnshift[o1];
    v0 = v0 > 0.f ? v0 : 0.2f * v0;
    v1 = v1 > 0.f ? v1 : 0.2f * v1;
    out[((size_t)(b * O + o0)) * Ncols + col] = v0;
    out[((size_t)(b * O + o1)) * Ncols + col] = v1;
  }
}

// ---------------------------------------------------------------------------
// CBAM pieces
// ---------------------------------------------------------------------------
__global__ __launch_bounds__(256) void ca_reduce_kernel(
    const float* __restrict__ f, float* __restrict__ camean,
    float* __restrict__ camax) {
  int bo = blockIdx.x;
  const float* p = f + (size_t)bo * NKCOLS;
  float s = 0.f, mx = NEG_INF;
  for (int i = threadIdx.x; i < NKCOLS; i += 256) {
    float v = p[i]; s += v; mx = fmaxf(mx, v);
  }
  __shared__ float ss[256], sm[256];
  ss[threadIdx.x] = s; sm[threadIdx.x] = mx;
  __syncthreads();
  for (int o = 128; o > 0; o >>= 1) {
    if (threadIdx.x < o) {
      ss[threadIdx.x] += ss[threadIdx.x + o];
      sm[threadIdx.x] = fmaxf(sm[threadIdx.x], sm[threadIdx.x + o]);
    }
    __syncthreads();
  }
  if (threadIdx.x == 0) {
    camean[bo] = ss[0] / (float)NKCOLS;
    camax[bo]  = sm[0];
  }
}

__global__ __launch_bounds__(256) void ca_mlp_kernel(
    const float* __restrict__ camean, const float* __restrict__ camax,
    const float* __restrict__ fc1, const float* __restrict__ fc2,
    float* __restrict__ ca, int O, int r) {
  int b = blockIdx.x, tid = threadIdx.x;
  __shared__ float sme[256], sma[256], hm[64], hx[64];
  if (tid < O) { sme[tid] = camean[b*O + tid]; sma[tid] = camax[b*O + tid]; }
  __syncthreads();
  if (tid < r) {
    float am = 0.f, ax = 0.f;
    for (int q = 0; q < O; ++q) {
      float w = fc1[tid*O + q];
      am += w * sme[q]; ax += w * sma[q];
    }
    hm[tid] = fmaxf(am, 0.f); hx[tid] = fmaxf(ax, 0.f);
  }
  __syncthreads();
  if (tid < O) {
    float s = 0.f;
    for (int q = 0; q < r; ++q) s += fc2[tid*r + q] * (hm[q] + hx[q]);
    ca[b*O + tid] = 1.f / (1.f + __expf(-s));
  }
}

__global__ __launch_bounds__(256) void spatial_stats_kernel(
    const float* __restrict__ f, const float* __restrict__ ca,
    float* __restrict__ smean, float* __restrict__ smax, int O) {
  int i = blockIdx.x * blockDim.x + threadIdx.x;
  if (i >= NB * NKCOLS) return;
  int b = i / NKCOLS, p = i % NKCOLS;
  const float* fb = f + (size_t)b * O * NKCOLS + p;
  const float* cb = ca + b * O;
  float s = 0.f, mx = NEG_INF;
  for (int o = 0; o < O; ++o) {
    float v = fb[(size_t)o * NKCOLS] * cb[o];
    s += v; mx = fmaxf(mx, v);
  }
  smean[i] = s / (float)O;
  smax[i]  = mx;
}

__global__ __launch_bounds__(256) void spatial_conv_kernel(
    const float* __restrict__ smean, const float* __restrict__ smax,
    const float* __restrict__ sw, float* __restrict__ sa) {
  int i = blockIdx.x * blockDim.x + threadIdx.x;
  if (i >= NB * NKCOLS) return;
  int b = i / NKCOLS, p = i % NKCOLS;
  int n = p / KNN, kk = p % KNN;
  const float* me = smean + (size_t)b * NKCOLS;
  const float* mx = smax  + (size_t)b * NKCOLS;
  float acc = 0.f;
#pragma unroll
  for (int dh = 0; dh < 5; ++dh) {
    int nn = n + dh - 2;
    if (nn < 0 || nn >= NPTS) continue;
#pragma unroll
    for (int dw = 0; dw < 5; ++dw) {
      int kw = kk + dw - 2;
      if (kw < 0 || kw >= KNN) continue;
      int pp = nn * KNN + kw;
      acc += sw[dh*5 + dw] * me[pp] + sw[25 + dh*5 + dw] * mx[pp];
    }
  }
  sa[i] = 1.f / (1.f + __expf(-acc));
}

__global__ __launch_bounds__(256) void kmax_kernel(
    const float* __restrict__ f, const float* __restrict__ ca,
    const float* __restrict__ sa, float* __restrict__ xcat,
    int O, int rowoff) {
  int i = blockIdx.x * blockDim.x + threadIdx.x;
  if (i >= NB * O * NPTS) return;
  int b = i / (O * NPTS);
  int rm = i % (O * NPTS);
  int o = rm / NPTS, n = rm % NPTS;
  const float* fp = f + ((size_t)(b * O + o)) * NKCOLS + (size_t)n * KNN;
  const float* sp = sa + (size_t)b * NKCOLS + (size_t)n * KNN;
  float c = ca[b*O + o];
  float mx = NEG_INF;
#pragma unroll
  for (int kk = 0; kk < KNN; ++kk) mx = fmaxf(mx, fp[kk] * c * sp[kk]);
  xcat[((size_t)b * 512 + rowoff + o) * NPTS + n] = mx;
}

// Global max+mean pooling over N per (b,o): g[b][0..1023]=max, [1024..2047]=mean
__global__ __launch_bounds__(256) void pool_kernel(
    const float* __restrict__ hbuf, float* __restrict__ g) {
  int bo = blockIdx.x;              // b*1024 + o
  const float* p = hbuf + (size_t)bo * NPTS;
  float s = 0.f, mx = NEG_INF;
  for (int i = threadIdx.x; i < NPTS; i += 256) {
    float v = p[i]; s += v; mx = fmaxf(mx, v);
  }
  __shared__ float ss[256], sm[256];
  ss[threadIdx.x] = s; sm[threadIdx.x] = mx;
  __syncthreads();
  for (int o = 128; o > 0; o >>= 1) {
    if (threadIdx.x < o) {
      ss[threadIdx.x] += ss[threadIdx.x + o];
      sm[threadIdx.x] = fmaxf(sm[threadIdx.x], sm[threadIdx.x + o]);
    }
    __syncthreads();
  }
  if (threadIdx.x == 0) {
    int b = bo / 1024, o = bo % 1024;
    g[(size_t)b * 2048 + o]        = sm[0];
    g[(size_t)b * 2048 + 1024 + o] = ss[0] / (float)NPTS;
  }
}

// MLP head: 2048 -> 512(bn6,leaky) -> 256(bn7,leaky) -> 40
__global__ __launch_bounds__(256) void head_kernel(
    const float* __restrict__ g,  const float* __restrict__ lin1,
    const float* __restrict__ s6, const float* __restrict__ h6,
    const float* __restrict__ lin2, const float* __restrict__ l2b,
    const float* __restrict__ s7, const float* __restrict__ h7,
    const float* __restrict__ lin3, const float* __restrict__ l3b,
    float* __restrict__ out) {
  int b = blockIdx.x, tid = threadIdx.x;
  __shared__ float gv[2048], h1[512], h2[256];
  for (int i = tid; i < 2048; i += 256) gv[i] = g[(size_t)b * 2048 + i];
  __syncthreads();
  for (int row = tid; row < 512; row += 256) {
    const float* w = lin1 + (size_t)row * 2048;
    float s = 0.f;
    for (int c = 0; c < 2048; ++c) s += w[c] * gv[c];
    s = s * s6[row] + h6[row];
    h1[row] = s > 0.f ? s : 0.2f * s;
  }
  __syncthreads();
  if (tid < 256) {
    const float* w = lin2 + (size_t)tid * 512;
    float s = 0.f;
    for (int c = 0; c < 512; ++c) s += w[c] * h1[c];
    s += l2b[tid];
    s = s * s7[tid] + h7[tid];
    h2[tid] = s > 0.f ? s : 0.2f * s;
  }
  __syncthreads();
  if (tid < 40) {
    const float* w = lin3 + (size_t)tid * 256;
    float s = 0.f;
    for (int c = 0; c < 256; ++c) s += w[c] * h2[c];
    out[b * 40 + tid] = s + l3b[tid];
  }
}

// ---------------------------------------------------------------------------
// Host orchestration
// ---------------------------------------------------------------------------
extern "C" void kernel_launch(void* const* d_in, const int* in_sizes, int n_in,
                              void* d_out, int out_size, void* d_ws, size_t ws_size,
                              hipStream_t stream) {
  (void)in_sizes; (void)n_in; (void)out_size; (void)ws_size;
  const float* x    = (const float*)d_in[0];
  const float* Ws[4]  = {(const float*)d_in[1], (const float*)d_in[2],
                         (const float*)d_in[3], (const float*)d_in[4]};
  const float* W5   = (const float*)d_in[5];
  const float* bns[4] = {(const float*)d_in[6], (const float*)d_in[7],
                         (const float*)d_in[8], (const float*)d_in[9]};
  const float* bn5  = (const float*)d_in[10];
  const float* bn6  = (const float*)d_in[11];
  const float* bn7  = (const float*)d_in[12];
  const float* fc1s[4] = {(const float*)d_in[13], (const float*)d_in[16],
                          (const float*)d_in[19], (const float*)d_in[22]};
  const float* fc2s[4] = {(const float*)d_in[14], (const float*)d_in[17],
                          (const float*)d_in[20], (const float*)d_in[23]};
  const float* saws[4] = {(const float*)d_in[15], (const float*)d_in[18],
                          (const float*)d_in[21], (const float*)d_in[24]};
  const float* lin1 = (const float*)d_in[25];
  const float* lin2 = (const float*)d_in[26];
  const float* l2b  = (const float*)d_in[27];
  const float* lin3 = (const float*)d_in[28];
  const float* l3b  = (const float*)d_in[29];

  // workspace carve-out (256B aligned)
  char* base = (char*)d_ws;
  size_t off = 0;
  auto carve = [&](size_t bytes) -> void* {
    void* p = base + off;
    off += (bytes + 255) & ~(size_t)255;
    return p;
  };
  float* xcat   = (float*)carve((size_t)NB * 512 * NPTS * 4);        // 16.8 MB
  float* fbuf   = (float*)carve((size_t)NB * 256 * NKCOLS * 4);      // 168 MB
  float* pdbuf  = (float*)carve((size_t)NB * NPTS * NPTS * 4);       // 67 MB (also h)
  f16*   xhT    = (f16*)  carve((size_t)NB * NPTS * 128 * 2);        // point-major
  f16*   xcatT  = (f16*)  carve((size_t)NB * NPTS * 512 * 2);        // 8.4 MB
  float* xx     = (float*)carve((size_t)NB * NPTS * 4);
  int*   idxb   = (int*)  carve((size_t)NB * NPTS * KNN * 4);
  f16*   Wh     = (f16*)  carve((size_t)1024 * 512 * 2);
  float* bnsc   = (float*)carve(1024 * 4);
  float* bnsh   = (float*)carve(1024 * 4);
  float* camean = (float*)carve(1024 * 4);
  float* camax_ = (float*)carve(1024 * 4);
  float* cav    = (float*)carve(1024 * 4);
  float* smean  = (float*)carve((size_t)NB * NKCOLS * 4);
  float* smaxb  = (float*)carve((size_t)NB * NKCOLS * 4);
  float* sabuf  = (float*)carve((size_t)NB * NKCOLS * 4);
  float* gbuf   = (float*)carve((size_t)NB * 2048 * 4);
  float* s6 = (float*)carve(512 * 4);  float* h6 = (float*)carve(512 * 4);
  float* s7 = (float*)carve(256 * 4);  float* h7 = (float*)carve(256 * 4);

  const int Cin[4]  = {3, 64, 64, 128};
  const int Cpad[4] = {32, 64, 64, 128};
  const int Oc[4]   = {64, 64, 128, 256};
  const int Kpad[4] = {32, 128, 128, 256};
  const int Cw[4]   = {6, 128, 128, 256};
  const int rowoff[4] = {0, 64, 128, 256};
  const int rr[4]   = {8, 8, 16, 32};
  const float* xin0[4] = {x, xcat + 0 * NPTS, xcat + 64 * (size_t)NPTS,
                          xcat + 128 * (size_t)NPTS};
  const long bstride[4] = {3L * NPTS, 512L * NPTS, 512L * NPTS, 512L * NPTS};

  for (int bl = 0; bl < 4; ++bl) {
    int O = Oc[bl];
    cvt_w_kernel<<<cdiv(O * Kpad[bl], 256), 256, 0, stream>>>(
        Ws[bl], Wh, O, Cw[bl], Kpad[bl]);
    bn_prep_kernel<<<cdiv(O, 256), 256, 0, stream>>>(bns[bl], bnsc, bnsh, O);
    cvt_xh_kernel<<<cdiv(NB * NPTS * Cpad[bl], 256), 256, 0, stream>>>(
        xin0[bl], bstride[bl], xhT, Cin[bl], Cpad[bl]);
    sqnorm_kernel<<<cdiv(NB * NPTS, 256), 256, 0, stream>>>(
        xin0[bl], bstride[bl], xx, Cin[bl]);
    gram_pd_kernel<<<NB * 128 * 64 / 4, 128, 0, stream>>>(xhT, xx, pdbuf, Cpad[bl]);
    topk_kernel<<<NB * NPTS / 4, 128, 0, stream>>>(pdbuf, idxb);
    edge_gemm_kernel<<<NB * (O / 32) * (NKCOLS / 16) / 4, 128, 0, stream>>>(
        Wh, xhT, Cpad[bl], idxb, bnsc, bnsh, fbuf, O, Cin[bl], Kpad[bl]);
    ca_reduce_kernel<<<NB * O, 256, 0, stream>>>(fbuf, camean, camax_);
    ca_mlp_kernel<<<NB, 256, 0, stream>>>(camean, camax_, fc1s[bl], fc2s[bl],
                                          cav, O, rr[bl]);
    spatial_stats_kernel<<<cdiv(NB * NKCOLS, 256), 256, 0, stream>>>(
        fbuf, cav, smean, smaxb, O);
    spatial_conv_kernel<<<cdiv(NB * NKCOLS, 256), 256, 0, stream>>>(
        smean, smaxb, saws[bl], sabuf);
    kmax_kernel<<<cdiv(NB * O * NPTS, 256), 256, 0, stream>>>(
        fbuf, cav, sabuf, xcat, O, rowoff[bl]);
  }

  // W5: 1024 x 512 conv over xcat (point-major f16), BN5 + leaky, then pool
  cvt_w_kernel<<<cdiv(1024 * 512, 256), 256, 0, stream>>>(W5, Wh, 1024, 512, 512);
  bn_prep_kernel<<<cdiv(1024, 256), 256, 0, stream>>>(bn5, bnsc, bnsh, 1024);
  cvt_xcatT_kernel<<<cdiv(NB * NPTS * 512, 256), 256, 0, stream>>>(xcat, xcatT);
  plain_gemm_tdm_kernel<<<NB * (1024 / 32) * (NPTS / 16 / 4), 128, 0, stream>>>(
      Wh, xcatT, bnsc, bnsh, pdbuf);
  pool_kernel<<<NB * 1024, 256, 0, stream>>>(pdbuf, gbuf);

  bn_prep_kernel<<<cdiv(512, 256), 256, 0, stream>>>(bn6, s6, h6, 512);
  bn_prep_kernel<<<1, 256, 0, stream>>>(bn7, s7, h7, 256);
  head_kernel<<<NB, 256, 0, stream>>>(gbuf, lin1, s6, h6, lin2, l2b, s7, h7,
                                      lin3, l3b, (float*)d_out);
}
